// SorscherRNN_56899726737884
// MI455X (gfx1250) — matrix operand order repaired and checked by reading, not compiled, not run on hardware
//
#include <hip/hip_runtime.h>
#include <hip/hip_bf16.h>

typedef __bf16 v16bf __attribute__((ext_vector_type(16)));
typedef __bf16 v8bf  __attribute__((ext_vector_type(8)));
typedef float  v8f   __attribute__((ext_vector_type(8)));

#define B_  64
#define S_  128
#define NG  4096
#define NP  512

// ---------------- fp32 -> bf16 weight conversion ----------------
__global__ __launch_bounds__(256)
void cvt_f32_bf16(const float* __restrict__ src, __bf16* __restrict__ dst, int n) {
  int i = blockIdx.x * 256 + threadIdx.x;
  if (i < n) dst[i] = (__bf16)src[i];
}

// ---------------- h0 = p0 @ W_init^T (small, fp32 dot) ----------------
__global__ __launch_bounds__(256)
void h0_kernel(const float* __restrict__ p0, const float* __restrict__ Winit,
               __bf16* __restrict__ h0) {
  int idx = blockIdx.x * 256 + threadIdx.x;   // B_*NG threads
  int b = idx >> 12;        // /4096
  int g = idx & (NG - 1);
  const float* pr = p0 + b * NP;
  const float* wr = Winit + g * NP;
  float acc = 0.f;
  #pragma unroll 4
  for (int k = 0; k < NP; k += 4) {
    acc += pr[k+0]*wr[k+0] + pr[k+1]*wr[k+1] + pr[k+2]*wr[k+2] + pr[k+3]*wr[k+3];
  }
  h0[idx] = (__bf16)acc;
}

// ---------------- one recurrence step: h = relu(vx_s + hprev @ Whh^T) ----------------
// 256 threads = 8 waves. Each wave computes a 16x32 output tile (2 accumulators),
// so one A fragment feeds two WMMAs. WG covers 64x64; grid.x = NG/64 = 64.
__global__ __launch_bounds__(256)
void rnn_step(const __bf16* __restrict__ Whh,    // [NG][NG] row-major (row g, col k)
              const __bf16* __restrict__ hprev,  // [B_][NG] bf16
              __bf16* __restrict__ hout,         // [B_][NG] bf16
              const float* __restrict__ v,       // [B_][S_][2]
              const float* __restrict__ Wih,     // [NG][2]
              int s) {
  const int lane = threadIdx.x & 31;
  const int wave = threadIdx.x >> 5;
  const int m0   = (wave & 3) * 16;                               // batch tile
  const int n0   = blockIdx.x * 64 + (wave >> 2) * 32;            // 32-wide Ng slab
  const int row  = lane & 15;                                     // A: M row / B: N col
  const int half = lane >> 4;

  // A (16x32, this code's contiguous k order; B uses the same): lanes 0-15 hold K {0..15}, lanes 16-31 hold K {16..31}
  const __bf16* aBase  = hprev + (size_t)(m0 + row) * NG + half * 16;
  // B (32x16): lane n holds 16 contiguous K; B[k][n] = Whh[n][k] (row-major row = WMMA column)
  const __bf16* bBase0 = Whh + (size_t)(n0 + row)      * NG + half * 16;
  const __bf16* bBase1 = Whh + (size_t)(n0 + 16 + row) * NG + half * 16;

  v8f acc0 = {};
  v8f acc1 = {};
  #pragma unroll 4
  for (int k0 = 0; k0 < NG; k0 += 32) {
    v8bf alo = *(const v8bf*)(aBase + k0);
    v8bf ahi = *(const v8bf*)(aBase + k0 + 8);
    v16bf a = __builtin_shufflevector(alo, ahi, 0,1,2,3,4,5,6,7,8,9,10,11,12,13,14,15);
    v16bf b0 = *(const v16bf*)(bBase0 + k0);
    v16bf b1 = *(const v16bf*)(bBase1 + k0);
    __builtin_prefetch(bBase0 + k0 + 256, 0, 1);   // global_prefetch_b8 down the B streams
    __builtin_prefetch(bBase1 + k0 + 256, 0, 1);
    acc0 = __builtin_amdgcn_wmma_f32_16x16x32_bf16(false, a, false, b0, (short)0, acc0,
                                                   false, false);
    acc1 = __builtin_amdgcn_wmma_f32_16x16x32_bf16(false, a, false, b1, (short)0, acc1,
                                                   false, false);
  }

  // epilogue: add input projection, relu, store bf16
  const int g0 = n0 + row;
  const int g1 = g0 + 16;
  const float wi00 = Wih[g0 * 2 + 0], wi01 = Wih[g0 * 2 + 1];
  const float wi10 = Wih[g1 * 2 + 0], wi11 = Wih[g1 * 2 + 1];
  #pragma unroll
  for (int r = 0; r < 8; ++r) {
    const int m = m0 + r + half * 8;                 // batch index (C/D layout)
    const float vv0 = v[((size_t)m * S_ + s) * 2 + 0];
    const float vv1 = v[((size_t)m * S_ + s) * 2 + 1];
    float h0v = acc0[r] + vv0 * wi00 + vv1 * wi01;
    float h1v = acc1[r] + vv0 * wi10 + vv1 * wi11;
    h0v = h0v > 0.f ? h0v : 0.f;
    h1v = h1v > 0.f ? h1v : 0.f;
    hout[(size_t)m * NG + g0] = (__bf16)h0v;
    hout[(size_t)m * NG + g1] = (__bf16)h1v;
  }
}

// ---------------- decoder: out[b,s,p] = sum_g hs[s,b,g] * Wdec[p,g] ----------------
// M = S_*B_ = 8192 rows (r = s*64 + b), N = NP = 512, K = NG.
// Each wave computes a 16x32 tile: 512 m-tiles x 16 n-slabs = 8192 wave-tiles, 1024 WGs.
__global__ __launch_bounds__(256)
void decoder_kernel(const __bf16* __restrict__ hs,    // rows r = s*B_ + b, [8192][NG]
                    const __bf16* __restrict__ Wdec,  // [NP][NG]
                    float* __restrict__ out) {        // [B_][S_][NP]
  const int lane = threadIdx.x & 31;
  const int wave = threadIdx.x >> 5;
  const int t = blockIdx.x * 8 + wave;   // 8192 wave-tiles total
  const int n_slab = t & 15;             // NP/32 = 16
  const int m_tile = t >> 4;             // 8192/16 = 512
  const int m0 = m_tile * 16, n0 = n_slab * 32;
  const int row = lane & 15, half = lane >> 4;

  const __bf16* aBase  = hs   + (size_t)(m0 + row) * NG + half * 16;
  const __bf16* bBase0 = Wdec + (size_t)(n0 + row)      * NG + half * 16;
  const __bf16* bBase1 = Wdec + (size_t)(n0 + 16 + row) * NG + half * 16;

  v8f acc0 = {};
  v8f acc1 = {};
  #pragma unroll 4
  for (int k0 = 0; k0 < NG; k0 += 32) {
    v8bf alo = *(const v8bf*)(aBase + k0);
    v8bf ahi = *(const v8bf*)(aBase + k0 + 8);
    v16bf a = __builtin_shufflevector(alo, ahi, 0,1,2,3,4,5,6,7,8,9,10,11,12,13,14,15);
    v16bf b0 = *(const v16bf*)(bBase0 + k0);
    v16bf b1 = *(const v16bf*)(bBase1 + k0);
    acc0 = __builtin_amdgcn_wmma_f32_16x16x32_bf16(false, a, false, b0, (short)0, acc0,
                                                   false, false);
    acc1 = __builtin_amdgcn_wmma_f32_16x16x32_bf16(false, a, false, b1, (short)0, acc1,
                                                   false, false);
  }

  const int p0 = n0 + row;
  const int p1 = p0 + 16;
  #pragma unroll
  for (int r = 0; r < 8; ++r) {
    const int mr = m0 + r + half * 8;    // mr = s*64 + b
    const int b  = mr & 63;
    const int sIdx = mr >> 6;
    out[((size_t)b * S_ + sIdx) * NP + p0] = acc0[r];
    out[((size_t)b * S_ + sIdx) * NP + p1] = acc1[r];
  }
}

extern "C" void kernel_launch(void* const* d_in, const int* in_sizes, int n_in,
                              void* d_out, int out_size, void* d_ws, size_t ws_size,
                              hipStream_t stream) {
  (void)in_sizes; (void)n_in; (void)out_size; (void)ws_size;
  const float* v     = (const float*)d_in[0];  // [64,128,2]
  const float* p0    = (const float*)d_in[1];  // [64,512]
  const float* Winit = (const float*)d_in[2];  // [4096,512]
  const float* Wih   = (const float*)d_in[3];  // [4096,2]
  const float* Whh   = (const float*)d_in[4];  // [4096,4096]
  const float* Wdec  = (const float*)d_in[5];  // [512,4096]
  float* out = (float*)d_out;

  // workspace layout (bytes): Whh_bf16 | Wdec_bf16 | h_seq[(S_+1)][B_][NG] bf16
  char* ws = (char*)d_ws;
  __bf16* Whh_bf  = (__bf16*)ws;
  __bf16* Wdec_bf = (__bf16*)(ws + (size_t)NG * NG * 2);
  __bf16* hseq    = (__bf16*)(ws + (size_t)NG * NG * 2 + (size_t)NP * NG * 2);

  // 1) convert weights to bf16 (Whh stays L2-resident: 32 MB << 192 MB)
  cvt_f32_bf16<<<(NG * NG + 255) / 256, 256, 0, stream>>>(Whh, Whh_bf, NG * NG);
  cvt_f32_bf16<<<(NP * NG + 255) / 256, 256, 0, stream>>>(Wdec, Wdec_bf, NP * NG);

  // 2) h0
  h0_kernel<<<(B_ * NG) / 256, 256, 0, stream>>>(p0, Winit, hseq);

  // 3) 128 serial recurrence steps, dependency via stream ordering (graph-captured)
  for (int s = 0; s < S_; ++s) {
    rnn_step<<<NG / 64, 256, 0, stream>>>(Whh_bf,
                                          hseq + (size_t)s       * B_ * NG,
                                          hseq + (size_t)(s + 1) * B_ * NG,
                                          v, Wih, s);
  }

  // 4) decoder over all (s,b) rows
  decoder_kernel<<<8192 / 8, 256, 0, stream>>>(hseq + (size_t)B_ * NG, Wdec_bf, out);
}
